// MultiHeadAttention_55920474193939
// MI455X (gfx1250) — matrix operand order repaired and checked by reading, not compiled, not run on hardware
//
#include <hip/hip_runtime.h>
#include <hip/hip_bf16.h>
#include <math.h>

// Problem constants (match reference)
#define B_   16
#define S_   512
#define DM_  512
#define H_   8
#define DK_  64
#define BH_  (B_*H_)        // 128
#define ROWS (B_*S_)        // 8192
#define LDB  72             // LDS tile stride in halves (144 B, 16B-aligned)

typedef __attribute__((ext_vector_type(16))) _Float16 v16h;
typedef __attribute__((ext_vector_type(8)))  float    v8f;

union FragU { v16h v; uint4 q[2]; };
union H4   { _Float16 h[4]; unsigned long long u; };

// ---- WMMA fragment loaders (ISA 7.12.2 layouts, wave32) -------------------
// A (16x32 f16): lane L holds row (L&15); halves 0..7 = k0+koff+0..7,
// halves 8..15 = k0+16+koff+0..7, koff = 8*(L>=16).
__device__ __forceinline__ v16h load_fragA(const _Float16* a, int ld,
                                           int rowBase, int k0, int lane) {
    int r    = rowBase + (lane & 15);
    int koff = (lane >= 16) ? 8 : 0;
    const _Float16* p = a + (size_t)r * ld + k0 + koff;
    FragU f;
    f.q[0] = *(const uint4*)(p);        // k0+koff   .. +7
    f.q[1] = *(const uint4*)(p + 16);   // k0+16+koff.. +7
    return f.v;
}

// B (32x16 f16), source stored COLUMN-major as Bt[n][k] (row-major over n):
// lane L holds column (L&15); halves 0..7 = k0+koff+0..7,
// halves 8..15 = k0+16+koff+0..7, koff = 8*(L>=16).
__device__ __forceinline__ v16h load_fragB(const _Float16* bt, int ld,
                                           int colBase, int k0, int lane) {
    int c    = colBase + (lane & 15);
    int koff = (lane >= 16) ? 8 : 0;
    const _Float16* p = bt + (size_t)c * ld + k0 + koff;
    FragU f;
    f.q[0] = *(const uint4*)(p);        // 8 halves
    f.q[1] = *(const uint4*)(p + 16);   // k0+16+koff.. +7
    return f.v;
}

#define WMMA(a, b, c) __builtin_amdgcn_wmma_f32_16x16x32_f16( \
    false, (a), false, (b), (short)0, (c), false, false)

// Stage A tile (64 rows x 64 k) f32 -> f16 into As[64][LDB].
// Static trip count: 1024 float4 / 128 threads = 8 iterations, b128 loads,
// b64 LDS stores.
__device__ __forceinline__ void stage_A64(const float* __restrict__ X,
                                          int rowBase, int ldx, int k0,
                                          _Float16* As, int t) {
    #pragma unroll
    for (int it = 0; it < 8; ++it) {
        int idx = t + it * 128;          // 0..1023
        int r   = idx >> 4;              // 16 float4 per row
        int c4  = (idx & 15) * 4;
        float4 f = *(const float4*)(X + (size_t)(rowBase + r) * ldx + k0 + c4);
        H4 p;
        p.h[0] = (_Float16)f.x; p.h[1] = (_Float16)f.y;
        p.h[2] = (_Float16)f.z; p.h[3] = (_Float16)f.w;
        *(unsigned long long*)(As + r * LDB + c4) = p.u;
    }
}

// Stage B tile transposed: W[k0..+64][n0..+64] (f32, ldw) -> Bs[n][k] f16.
// 1024 float4 / 128 threads = 8 iterations, b128 loads, 4 b16 LDS stores.
__device__ __forceinline__ void stage_Bt64(const float* __restrict__ W,
                                           int k0, int ldw, int n0,
                                           _Float16* Bs, int t) {
    #pragma unroll
    for (int it = 0; it < 8; ++it) {
        int idx = t + it * 128;          // 0..1023
        int k   = idx >> 4;              // 16 float4 per k-row
        int n4  = (idx & 15) * 4;
        float4 f = *(const float4*)(W + (size_t)(k0 + k) * ldw + n0 + n4);
        Bs[(n4 + 0) * LDB + k] = (_Float16)f.x;
        Bs[(n4 + 1) * LDB + k] = (_Float16)f.y;
        Bs[(n4 + 2) * LDB + k] = (_Float16)f.z;
        Bs[(n4 + 3) * LDB + k] = (_Float16)f.w;
    }
}

// ---------------------------------------------------------------------------
// Kernel 1: fused QKV projection.  X(f32 8192x512) @ W(f32 512x512) -> f16.
// z=0 -> Qh[b][h][s][dk], z=1 -> Kh[b][h][s][dk], z=2 -> Vt[b][h][dv][s].
// Block: 128 thr (4 waves), 64x64 tile, K-tile depth 64 (8 WMMA per fill).
// ---------------------------------------------------------------------------
__global__ __launch_bounds__(128)
void k_proj(const float* __restrict__ Xq, const float* __restrict__ Xk,
            const float* __restrict__ Xv, const float* __restrict__ Wq,
            const float* __restrict__ Wk, const float* __restrict__ Wv,
            _Float16* __restrict__ Qh, _Float16* __restrict__ Kh,
            _Float16* __restrict__ Vt) {
    __shared__ _Float16 As[64 * LDB];
    __shared__ _Float16 Bs[64 * LDB];

    const int z = blockIdx.z;
    const float* X = (z == 0) ? Xq : (z == 1) ? Xk : Xv;
    const float* W = (z == 0) ? Wq : (z == 1) ? Wk : Wv;

    const int t = threadIdx.x, lane = t & 31, w = t >> 5;
    const int wm = (w >> 1) * 32, wn = (w & 1) * 32;
    const int M0 = blockIdx.y * 64, N0 = blockIdx.x * 64;

    v8f acc[2][2] = {};

    for (int k0 = 0; k0 < DM_; k0 += 64) {
        stage_A64(X, M0, DM_, k0, As, t);
        stage_Bt64(W, k0, DM_, N0, Bs, t);
        __syncthreads();
        #pragma unroll
        for (int kk = 0; kk < 64; kk += 32) {
            v16h a0 = load_fragA(As, LDB, wm,      kk, lane);
            v16h a1 = load_fragA(As, LDB, wm + 16, kk, lane);
            v16h b0 = load_fragB(Bs, LDB, wn,      kk, lane);
            v16h b1 = load_fragB(Bs, LDB, wn + 16, kk, lane);
            acc[0][0] = WMMA(a0, b0, acc[0][0]);
            acc[0][1] = WMMA(a0, b1, acc[0][1]);
            acc[1][0] = WMMA(a1, b0, acc[1][0]);
            acc[1][1] = WMMA(a1, b1, acc[1][1]);
        }
        __syncthreads();
    }

    #pragma unroll
    for (int i = 0; i < 2; i++)
    #pragma unroll
    for (int j = 0; j < 2; j++) {
        int colg  = N0 + wn + j * 16 + (lane & 15);
        int rbase = M0 + wm + i * 16 + ((lane >= 16) ? 8 : 0);
        int h = colg >> 6, cc = colg & 63;
        #pragma unroll
        for (int r = 0; r < 8; r++) {
            int rowg = rbase + r;
            int b = rowg >> 9, s = rowg & 511;
            _Float16 v = (_Float16)acc[i][j][r];
            if (z == 2)
                Vt[((size_t)(b * H_ + h) * DK_ + cc) * S_ + s] = v;
            else if (z == 0)
                Qh[((size_t)(b * H_ + h) * S_ + s) * DK_ + cc] = v;
            else
                Kh[((size_t)(b * H_ + h) * S_ + s) * DK_ + cc] = v;
        }
    }
}

// ---------------------------------------------------------------------------
// Kernel 2: scores = Q @ K^T * 1/sqrt(64) + res_att  (mask -> -1e9), f32 out.
// Per (b,h): A=Qh (512x64 f16), Bt=Kh (row n = key, contiguous dk).
// Fragments stream directly from global (f16 already), no LDS round-trip.
// ---------------------------------------------------------------------------
__global__ __launch_bounds__(128)
void k_scores(const _Float16* __restrict__ Qh, const _Float16* __restrict__ Kh,
              const float* __restrict__ res_att,
              const unsigned char* __restrict__ mask,
              float* __restrict__ scores) {
    const int bh = blockIdx.z;
    const _Float16* Qb = Qh + (size_t)bh * S_ * DK_;
    const _Float16* Kb = Kh + (size_t)bh * S_ * DK_;

    const int t = threadIdx.x, lane = t & 31, w = t >> 5;
    const int wm = (w >> 1) * 32, wn = (w & 1) * 32;
    const int M0 = blockIdx.y * 64, N0 = blockIdx.x * 64;

    v8f acc[2][2] = {};
    #pragma unroll
    for (int k0 = 0; k0 < DK_; k0 += 32) {
        v16h a0 = load_fragA(Qb, DK_, M0 + wm,      k0, lane);
        v16h a1 = load_fragA(Qb, DK_, M0 + wm + 16, k0, lane);
        v16h b0 = load_fragB(Kb, DK_, N0 + wn,      k0, lane);
        v16h b1 = load_fragB(Kb, DK_, N0 + wn + 16, k0, lane);
        acc[0][0] = WMMA(a0, b0, acc[0][0]);
        acc[0][1] = WMMA(a0, b1, acc[0][1]);
        acc[1][0] = WMMA(a1, b0, acc[1][0]);
        acc[1][1] = WMMA(a1, b1, acc[1][1]);
    }

    const float inv = 0.125f;  // 1/sqrt(DK)
    const int b = bh >> 3;
    const size_t base = (size_t)bh * S_ * S_;
    #pragma unroll
    for (int i = 0; i < 2; i++)
    #pragma unroll
    for (int j = 0; j < 2; j++) {
        int kcol = N0 + wn + j * 16 + (lane & 15);
        int q0   = M0 + wm + i * 16 + ((lane >= 16) ? 8 : 0);
        #pragma unroll
        for (int r = 0; r < 8; r++) {
            int q = q0 + r;
            size_t off = base + (size_t)q * S_ + kcol;
            float v = acc[i][j][r] * inv + res_att[off];
            size_t moff = ((size_t)b * S_ + q) * S_ + kcol;
            if (mask[moff]) v = -1e9f;
            scores[off] = v;
        }
    }
}

// ---------------------------------------------------------------------------
// Kernel 3: softmax over the QUERY axis (axis=3 of [b,d,h,q,k]) — faithful to
// reference: each thread owns one column k, normalizes over the 512 q values.
// Adjacent threads own adjacent k -> coalesced; column walk hits L2 (192 MB).
// ---------------------------------------------------------------------------
__global__ __launch_bounds__(256)
void k_softmax(const float* __restrict__ scores, _Float16* __restrict__ attn) {
    int t  = blockIdx.x * blockDim.x + threadIdx.x;   // 0 .. BH*S-1
    int bh = t >> 9, k = t & 511;
    const float* col = scores + (size_t)bh * S_ * S_ + k;
    float m = -INFINITY;
    #pragma unroll 8
    for (int q = 0; q < S_; q++) m = fmaxf(m, col[(size_t)q * S_]);
    float s = 0.f;
    #pragma unroll 8
    for (int q = 0; q < S_; q++) s += __expf(col[(size_t)q * S_] - m);
    float inv = 1.f / s;
    _Float16* acol = attn + (size_t)bh * S_ * S_ + k;
    #pragma unroll 8
    for (int q = 0; q < S_; q++)
        acol[(size_t)q * S_] = (_Float16)(__expf(col[(size_t)q * S_] - m) * inv);
}

// ---------------------------------------------------------------------------
// Kernel 4: context = attn @ V.  A=attn (512x512 f16), Bt=Vt (dv-major).
// Output f16 interleaved ctx[b][s][h*64+dv] for the final projection.
// ---------------------------------------------------------------------------
__global__ __launch_bounds__(128)
void k_context(const _Float16* __restrict__ attn,
               const _Float16* __restrict__ Vt, _Float16* __restrict__ ctx) {
    const int bh = blockIdx.z, b = bh >> 3, h = bh & 7;
    const _Float16* A  = attn + (size_t)bh * S_ * S_;
    const _Float16* Bt = Vt   + (size_t)bh * DK_ * S_;

    const int t = threadIdx.x, lane = t & 31, w = t >> 5;
    const int wm = (w >> 1) * 32, wn = (w & 1) * 32;
    const int M0 = blockIdx.y * 64, N0 = blockIdx.x * 64;

    v8f acc[2][2] = {};
    #pragma unroll 4
    for (int k0 = 0; k0 < S_; k0 += 32) {
        v16h a0 = load_fragA(A,  S_, M0 + wm,      k0, lane);
        v16h a1 = load_fragA(A,  S_, M0 + wm + 16, k0, lane);
        v16h b0 = load_fragB(Bt, S_, N0 + wn,      k0, lane);
        v16h b1 = load_fragB(Bt, S_, N0 + wn + 16, k0, lane);
        acc[0][0] = WMMA(a0, b0, acc[0][0]);
        acc[0][1] = WMMA(a0, b1, acc[0][1]);
        acc[1][0] = WMMA(a1, b0, acc[1][0]);
        acc[1][1] = WMMA(a1, b1, acc[1][1]);
    }

    #pragma unroll
    for (int i = 0; i < 2; i++)
    #pragma unroll
    for (int j = 0; j < 2; j++) {
        int dv = N0 + wn + j * 16 + (lane & 15);
        int q0 = M0 + wm + i * 16 + ((lane >= 16) ? 8 : 0);
        #pragma unroll
        for (int r = 0; r < 8; r++) {
            int q = q0 + r;
            ctx[((size_t)(b * S_ + q)) * DM_ + h * DK_ + dv] =
                (_Float16)acc[i][j][r];
        }
    }
}

// ---------------------------------------------------------------------------
// Kernel 5: out_pre = ctx @ W_fc + input_Q (residual).  W_fc f32 staged/
// transposed to f16 in LDS (K-tile 64); ctx fragments direct from global.
// ---------------------------------------------------------------------------
__global__ __launch_bounds__(128)
void k_outproj(const _Float16* __restrict__ ctx, const float* __restrict__ Wfc,
               const float* __restrict__ inQ, float* __restrict__ outpre) {
    __shared__ _Float16 Bs[64 * LDB];
    const int t = threadIdx.x, lane = t & 31, w = t >> 5;
    const int wm = (w >> 1) * 32, wn = (w & 1) * 32;
    const int M0 = blockIdx.y * 64, N0 = blockIdx.x * 64;

    v8f acc[2][2] = {};
    for (int k0 = 0; k0 < DM_; k0 += 64) {
        stage_Bt64(Wfc, k0, DM_, N0, Bs, t);
        __syncthreads();
        #pragma unroll
        for (int kk = 0; kk < 64; kk += 32) {
            v16h a0 = load_fragA(ctx, DM_, M0 + wm,      k0 + kk, lane);
            v16h a1 = load_fragA(ctx, DM_, M0 + wm + 16, k0 + kk, lane);
            v16h b0 = load_fragB(Bs, LDB, wn,      kk, lane);
            v16h b1 = load_fragB(Bs, LDB, wn + 16, kk, lane);
            acc[0][0] = WMMA(a0, b0, acc[0][0]);
            acc[0][1] = WMMA(a0, b1, acc[0][1]);
            acc[1][0] = WMMA(a1, b0, acc[1][0]);
            acc[1][1] = WMMA(a1, b1, acc[1][1]);
        }
        __syncthreads();
    }

    #pragma unroll
    for (int i = 0; i < 2; i++)
    #pragma unroll
    for (int j = 0; j < 2; j++) {
        int colg = N0 + wn + j * 16 + (lane & 15);
        int r0   = M0 + wm + i * 16 + ((lane >= 16) ? 8 : 0);
        #pragma unroll
        for (int r = 0; r < 8; r++) {
            size_t off = (size_t)(r0 + r) * DM_ + colg;
            outpre[off] = acc[i][j][r] + inQ[off];
        }
    }
}

// ---------------------------------------------------------------------------
// Kernel 6: LayerNorm over DM=512, one block (256 thr) per row.
// ---------------------------------------------------------------------------
__global__ __launch_bounds__(256)
void k_ln(const float* __restrict__ outpre, const float* __restrict__ gamma,
          const float* __restrict__ beta, float* __restrict__ out) {
    __shared__ float s1[256], s2[256];
    const int row = blockIdx.x, t = threadIdx.x;
    const float* x = outpre + (size_t)row * DM_;
    float a = x[t], b = x[t + 256];
    s1[t] = a + b;
    s2[t] = a * a + b * b;
    __syncthreads();
    for (int off = 128; off > 0; off >>= 1) {
        if (t < off) { s1[t] += s1[t + off]; s2[t] += s2[t + off]; }
        __syncthreads();
    }
    float mu   = s1[0] * (1.0f / DM_);
    float var  = s2[0] * (1.0f / DM_) - mu * mu;
    float rstd = rsqrtf(var + 1e-5f);
    out[(size_t)row * DM_ + t]       = (a - mu) * rstd * gamma[t]       + beta[t];
    out[(size_t)row * DM_ + t + 256] = (b - mu) * rstd * gamma[t + 256] + beta[t + 256];
}

// ---------------------------------------------------------------------------
extern "C" void kernel_launch(void* const* d_in, const int* in_sizes, int n_in,
                              void* d_out, int out_size, void* d_ws,
                              size_t ws_size, hipStream_t stream) {
    const float*         inQ   = (const float*)d_in[0];
    const float*         inK   = (const float*)d_in[1];
    const float*         inV   = (const float*)d_in[2];
    const unsigned char* mask  = (const unsigned char*)d_in[3];
    const float*         res   = (const float*)d_in[4];
    const float*         Wq    = (const float*)d_in[5];
    const float*         Wk    = (const float*)d_in[6];
    const float*         Wv    = (const float*)d_in[7];
    const float*         Wfc   = (const float*)d_in[8];
    const float*         gamma = (const float*)d_in[9];
    const float*         beta  = (const float*)d_in[10];

    float* out    = (float*)d_out;
    float* scores = out + (size_t)ROWS * DM_;   // second tuple output

    char* ws = (char*)d_ws;
    _Float16* attn   = (_Float16*)(ws);                         // 64 MB
    _Float16* Qh     = (_Float16*)(ws + 67108864ULL);           //  8 MB
    _Float16* Kh     = (_Float16*)(ws + 75497472ULL);           //  8 MB
    _Float16* Vt     = (_Float16*)(ws + 83886080ULL);           //  8 MB
    _Float16* ctx    = (_Float16*)(ws + 92274688ULL);           //  8 MB
    float*    outpre = (float*)   (ws + 100663296ULL);          // 16 MB

    k_proj   <<<dim3(8, 128, 3),  128, 0, stream>>>(inQ, inK, inV, Wq, Wk, Wv,
                                                    Qh, Kh, Vt);
    k_scores <<<dim3(8, 8, BH_),  128, 0, stream>>>(Qh, Kh, res, mask, scores);
    k_softmax<<<dim3(256),        256, 0, stream>>>(scores, attn);
    k_context<<<dim3(1, 8, BH_),  128, 0, stream>>>(attn, Vt, ctx);
    k_outproj<<<dim3(8, 128),     128, 0, stream>>>(ctx, Wfc, inQ, outpre);
    k_ln     <<<dim3(ROWS),       256, 0, stream>>>(outpre, gamma, beta, out);
}
